// PDNConvModel_83339545411647
// MI455X (gfx1250) — matrix operand order repaired and checked by reading, not compiled, not run on hardware
//
#include <hip/hip_runtime.h>

// ---------------- problem dims (match reference) ----------------
#define N_NODES 50000
#define N_EDGES 600000
#define G_GRAPHS 256
#define HID 128
#define EDIM 16
#define EH 64
#define NL 4
#define MIN_F 200
#define MH 256
#define MOUT 128
#define PH 512

typedef __bf16 bf16;
typedef __attribute__((ext_vector_type(16))) __bf16 v16bf;
typedef __attribute__((ext_vector_type(8)))  __bf16 v8bf;
typedef __attribute__((ext_vector_type(8)))  float  v8f;

__device__ __forceinline__ bf16 f2bf(float f) {
  union { float f; unsigned u; } a; a.f = f;
  unsigned r = a.u + 0x7FFFu + ((a.u >> 16) & 1u);   // round-to-nearest-even
  union { unsigned short s; bf16 b; } o;
  o.s = (unsigned short)(r >> 16);
  return o.b;
}

__device__ __forceinline__ void atomAddF(float* p, float v) {
  __hip_atomic_fetch_add(p, v, __ATOMIC_RELAXED, __HIP_MEMORY_SCOPE_AGENT);
}

// ---------------- utility kernels ----------------
__global__ void fill_kernel(float* p, float v, int n) {
  int i = blockIdx.x * blockDim.x + threadIdx.x;
  if (i < n) p[i] = v;
}

// Convert W[K,ncols] (row-major f32) into WMMA-B-fragment-native bf16 layout:
//   Wp[((ct*KC + kc)*32 + lane)*16 + e]  with col = ct*16 + (lane&15),
//   k = kc*32 + 8*(lane>>4) + (e&7) + 16*(e>>3)   (ISA 7.12.2 B layout for 16-bit 32x16)
// Each lane's 16 elements for one k-chunk are contiguous -> one 32B vector load.
__global__ void pack_w_kernel(const float* __restrict__ W, bf16* __restrict__ Wp,
                              int K, int ncols) {
  int idx = blockIdx.x * blockDim.x + threadIdx.x;
  if (idx >= K * ncols) return;
  int e    = idx & 15;
  int lane = (idx >> 4) & 31;
  int rest = idx >> 9;              // ct*KC + kc
  int KC   = K >> 5;
  int ct   = rest / KC;
  int kc   = rest - ct * KC;
  int col  = ct * 16 + (lane & 15);
  int k    = kc * 32 + 8 * (lane >> 4) + (e & 7) + 16 * (e >> 3);
  Wp[idx] = f2bf(W[(size_t)k * ncols + col]);
}

// ---------------- edge gate: sigmoid(relu(ea@W1+b1)@W2+b2) ----------------
__global__ void edge_gate_kernel(const float* __restrict__ edge_attr,
                                 const float* __restrict__ mW1,  // [EDIM,EH]
                                 const float* __restrict__ mb1,  // [EH]
                                 const float* __restrict__ mW2,  // [EH]
                                 const float* __restrict__ mb2,  // [1]
                                 float* __restrict__ wgate, int E) {
  __shared__ float sW1[EDIM * EH];
  __shared__ float sb1[EH];
  __shared__ float sW2[EH];
  int t = threadIdx.x;
  for (int i = t; i < EDIM * EH; i += blockDim.x) sW1[i] = mW1[i];
  if (t < EH) { sb1[t] = mb1[t]; sW2[t] = mW2[t]; }
  __syncthreads();
  int e = blockIdx.x * blockDim.x + t;
  if (e >= E) return;
  float ef[EDIM];
#pragma unroll
  for (int k = 0; k < EDIM; ++k) ef[k] = edge_attr[e * EDIM + k];
  float out = mb2[0];
  for (int j = 0; j < EH; ++j) {
    float h = sb1[j];
#pragma unroll
    for (int k = 0; k < EDIM; ++k) h += ef[k] * sW1[k * EH + j];
    h = fmaxf(h, 0.0f);
    out += h * sW2[j];
  }
  wgate[e] = 1.0f / (1.0f + __expf(-out));
}

__global__ void edge_deg_kernel(const int* __restrict__ ei, const float* __restrict__ wg,
                                float* __restrict__ deg, int E) {
  int e = blockIdx.x * blockDim.x + threadIdx.x;
  if (e < E) atomAddF(&deg[ei[E + e]], wg[e]);   // col index
}

__global__ void rsqrt_kernel(const float* __restrict__ deg, float* __restrict__ dinv, int n) {
  int i = blockIdx.x * blockDim.x + threadIdx.x;
  if (i < n) dinv[i] = rsqrtf(deg[i]);           // deg >= 1 (self loop)
}

// ---------------- WMMA GEMM ----------------
// C[rows,ncols] = op(A[rows,K](+a_bias) @ W[K,ncols] + bias) ; W pre-packed (pack_w_kernel).
// block = 256 = 8 waves; block covers 16 rows x 128 cols (wave -> 16-col tile).
// grid.x = rows/16, grid.y = ncols/128.  rows%16==0, K%32==0, ncols%128==0.
template <int K, bool ABIAS, bool BIAS, bool RELU>
__global__ __launch_bounds__(256)
void gemm_wmma_bf16(const float* __restrict__ A, const float* __restrict__ a_bias,
                    const bf16* __restrict__ Wp, const float* __restrict__ bias,
                    float* __restrict__ C, int ncols) {
  constexpr int KC = K / 32;
  __shared__ bf16 As[16 * K];
  const int t = threadIdx.x;
  const int row0 = blockIdx.x * 16;

  // stage A tile (16 x K) to LDS as bf16, optional broadcast bias fused
  constexpr int TOT = 16 * K;
#pragma unroll
  for (int i = t * 4; i < TOT; i += 256 * 4) {
    int r = i / K, k = i - r * K;
    const float4 v = *(const float4*)&A[(size_t)(row0 + r) * K + k];
    float b0 = 0.f, b1 = 0.f, b2 = 0.f, b3 = 0.f;
    if (ABIAS) { b0 = a_bias[k]; b1 = a_bias[k + 1]; b2 = a_bias[k + 2]; b3 = a_bias[k + 3]; }
    As[i + 0] = f2bf(v.x + b0);
    As[i + 1] = f2bf(v.y + b1);
    As[i + 2] = f2bf(v.z + b2);
    As[i + 3] = f2bf(v.w + b3);
  }
  __syncthreads();

  const int wave = t >> 5, lane = t & 31;
  const int half = lane >> 4, l16 = lane & 15;
  const int ct = blockIdx.y * 8 + wave;          // global 16-col tile index
  const int col0 = ct * 16;

  v8f acc = {};
#pragma unroll
  for (int kc = 0; kc < KC; ++kc) {
    // A frag: two contiguous 8-element (16B) LDS vectors
    const v8bf* ap = (const v8bf*)&As[l16 * K + kc * 32 + 8 * half];
    v8bf a0 = ap[0];                              // k = kc*32 + 8*half + 0..7
    v8bf a1 = ap[2];                              // k = kc*32 + 8*half + 16..23
    v16bf a = __builtin_shufflevector(a0, a1, 0, 1, 2, 3, 4, 5, 6, 7,
                                      8, 9, 10, 11, 12, 13, 14, 15);
    // B frag: one contiguous 32B vector from packed weights
    v16bf b = *(const v16bf*)&Wp[(((size_t)ct * KC + kc) * 32 + lane) * 16];
    acc = __builtin_amdgcn_wmma_f32_16x16x32_bf16(false, a, false, b, (short)0, acc,
                                                  false, false);
  }

  const int c = col0 + l16;
  float bv = 0.f;
  if (BIAS) bv = bias[c];
#pragma unroll
  for (int dv = 0; dv < 8; ++dv) {
    int r = row0 + dv + 8 * half;                // D: VGPR dv -> row dv (+8 upper half)
    float v = acc[dv] + bv;
    if (RELU) v = fmaxf(v, 0.0f);
    C[(size_t)r * ncols + c] = v;
  }
}

// ---------------- aggregation ----------------
__global__ void agg_init_kernel(const float* __restrict__ hl, const float* __restrict__ dinv,
                                float* __restrict__ agg, int total) {
  int i = blockIdx.x * blockDim.x + threadIdx.x;
  if (i < total) {
    int n = i >> 7;                               // /HID
    float di = dinv[n];
    agg[i] = hl[i] * di * di;                     // self-loop term (weight 1)
  }
}

// one wave per edge; lane moves 4 floats (b128 gather + 4 float atomics)
__global__ void edge_scatter_kernel(const int* __restrict__ ei, const float* __restrict__ wg,
                                    const float* __restrict__ dinv, const float* __restrict__ hl,
                                    float* __restrict__ agg, int E) {
  int wave = threadIdx.x >> 5, lane = threadIdx.x & 31;
  int e = blockIdx.x * 8 + wave;
  int r = ei[e], c = ei[E + e];
  float s = dinv[r] * wg[e] * dinv[c];
  const float4 v = ((const float4*)(hl + (size_t)r * HID))[lane];
  float* dst = agg + (size_t)c * HID + lane * 4;
  atomAddF(dst + 0, s * v.x);
  atomAddF(dst + 1, s * v.y);
  atomAddF(dst + 2, s * v.z);
  atomAddF(dst + 3, s * v.w);
}

// ---------------- mean pool ----------------
__global__ void pool_acc_kernel(const float* __restrict__ h, const int* __restrict__ bidx,
                                float* __restrict__ sums, float* __restrict__ cnt) {
  int n = blockIdx.x, t = threadIdx.x;
  int g = bidx[n];
  atomAddF(&sums[(size_t)g * HID + t], h[(size_t)n * HID + t]);
  if (t == 0) atomAddF(&cnt[g], 1.0f);
}

__global__ void pool_div_kernel(const float* __restrict__ sums, const float* __restrict__ cnt,
                                float* __restrict__ pred_in) {
  int g = blockIdx.x, t = threadIdx.x;
  pred_in[(size_t)g * (HID + MOUT) + t] = sums[(size_t)g * HID + t] / fmaxf(cnt[g], 1.0f);
}

// ---------------- small dense layers (G=256 rows, scalar path) ----------------
__global__ void linear_relu_kernel(const float* __restrict__ X, const float* __restrict__ W,
                                   const float* __restrict__ b, float* __restrict__ Y,
                                   int K, int ncols, int ystride, int yoff, int relu) {
  __shared__ float sx[512];
  int r = blockIdx.x, j = threadIdx.x;
  for (int i = j; i < K; i += blockDim.x) sx[i] = X[(size_t)r * K + i];
  __syncthreads();
  if (j < ncols) {
    float acc = b[j];
    for (int k = 0; k < K; ++k) acc += sx[k] * W[(size_t)k * ncols + j];
    if (relu) acc = fmaxf(acc, 0.0f);
    Y[(size_t)r * ystride + yoff + j] = acc;
  }
}

__global__ void final_out_kernel(const float* __restrict__ H, const float* __restrict__ w,
                                 const float* __restrict__ b, float* __restrict__ out) {
  __shared__ float red[64];
  int g = blockIdx.x, t = threadIdx.x;
  float acc = 0.0f;
  for (int k = t; k < PH; k += 64) acc += H[(size_t)g * PH + k] * w[k];
  red[t] = acc;
  __syncthreads();
  for (int s = 32; s > 0; s >>= 1) {
    if (t < s) red[t] += red[t + s];
    __syncthreads();
  }
  if (t == 0) out[g] = red[0] + b[0];
}

// ---------------- host side ----------------
extern "C" void kernel_launch(void* const* d_in, const int* in_sizes, int n_in,
                              void* d_out, int out_size, void* d_ws, size_t ws_size,
                              hipStream_t stream) {
  (void)in_sizes; (void)n_in; (void)out_size; (void)ws_size;
  const float* x          = (const float*)d_in[0];
  const int*   edge_index = (const int*)  d_in[1];
  const float* edge_attr  = (const float*)d_in[2];
  const int*   batch_idx  = (const int*)  d_in[3];
  const float* mol_feat   = (const float*)d_in[4];
  const float* conv_lin_W = (const float*)d_in[5];
  const float* conv_mW1   = (const float*)d_in[6];
  const float* conv_mb1   = (const float*)d_in[7];
  const float* conv_mW2   = (const float*)d_in[8];
  const float* conv_mb2   = (const float*)d_in[9];
  const float* conv_bias  = (const float*)d_in[10];
  const float* gcn_W      = (const float*)d_in[11];
  const float* gcn_b      = (const float*)d_in[12];
  const float* mlp_W0 = (const float*)d_in[13]; const float* mlp_b0 = (const float*)d_in[14];
  const float* mlp_W1 = (const float*)d_in[15]; const float* mlp_b1 = (const float*)d_in[16];
  const float* mlp_W2 = (const float*)d_in[17]; const float* mlp_b2 = (const float*)d_in[18];
  const float* mlp_W3 = (const float*)d_in[19]; const float* mlp_b3 = (const float*)d_in[20];
  const float* pred_W0 = (const float*)d_in[21]; const float* pred_b0 = (const float*)d_in[22];
  const float* pred_W1 = (const float*)d_in[23]; const float* pred_b1 = (const float*)d_in[24];
  const float* out_W = (const float*)d_in[25];   const float* out_b = (const float*)d_in[26];
  float* out = (float*)d_out;

  // bump allocator over workspace
  char* base = (char*)d_ws;
  size_t off = 0;
  auto alloc = [&](size_t bytes) -> void* {
    off = (off + 255) & ~(size_t)255;
    void* p = base + off;
    off += bytes;
    return p;
  };
  float* hbuf  = (float*)alloc((size_t)N_NODES * HID * 4);
  float* hl    = (float*)alloc((size_t)N_NODES * HID * 4);
  float* agg   = (float*)alloc((size_t)N_NODES * HID * 4);
  float* wgate = (float*)alloc((size_t)N_EDGES * 4);
  float* deg   = (float*)alloc((size_t)N_NODES * 4);
  float* dinv  = (float*)alloc((size_t)N_NODES * 4);
  float* sums  = (float*)alloc((size_t)G_GRAPHS * HID * 4);
  float* cnt   = (float*)alloc((size_t)G_GRAPHS * 4);
  float* pred_in = (float*)alloc((size_t)G_GRAPHS * (HID + MOUT) * 4);
  float* tmpA  = (float*)alloc((size_t)G_GRAPHS * MH * 4);
  float* tmpB  = (float*)alloc((size_t)G_GRAPHS * MH * 4);
  float* ph0   = (float*)alloc((size_t)G_GRAPHS * PH * 4);
  float* ph1   = (float*)alloc((size_t)G_GRAPHS * PH * 4);
  bf16* wl_bf  = (bf16*)alloc((size_t)NL * HID * HID * 2);
  bf16* wg_bf  = (bf16*)alloc((size_t)NL * HID * HID * 2);
  bf16* p0_bf  = (bf16*)alloc((size_t)(HID + MOUT) * PH * 2);
  bf16* p1_bf  = (bf16*)alloc((size_t)PH * PH * 2);

  // pack weights to fragment-native bf16 layout (idempotent, graph-safe)
  auto pack = [&](const float* s, bf16* d, int K, int ncols) {
    int n = K * ncols;
    pack_w_kernel<<<(n + 255) / 256, 256, 0, stream>>>(s, d, K, ncols);
  };
  for (int i = 0; i < NL; ++i) {
    pack(conv_lin_W + (size_t)i * HID * HID, wl_bf + (size_t)i * HID * HID, HID, HID);
    pack(gcn_W + (size_t)i * HID * HID,      wg_bf + (size_t)i * HID * HID, HID, HID);
  }
  pack(pred_W0, p0_bf, HID + MOUT, PH);
  pack(pred_W1, p1_bf, PH, PH);

  const dim3 gemmGridN(N_NODES / 16, 1);        // 3125 x 1  (ncols=128)
  const float* hin = x;
  for (int i = 0; i < NL; ++i) {
    edge_gate_kernel<<<(N_EDGES + 255) / 256, 256, 0, stream>>>(
        edge_attr, conv_mW1 + (size_t)i * EDIM * EH, conv_mb1 + (size_t)i * EH,
        conv_mW2 + (size_t)i * EH, conv_mb2 + i, wgate, N_EDGES);
    fill_kernel<<<(N_NODES + 255) / 256, 256, 0, stream>>>(deg, 1.0f, N_NODES);
    edge_deg_kernel<<<(N_EDGES + 255) / 256, 256, 0, stream>>>(edge_index, wgate, deg, N_EDGES);
    rsqrt_kernel<<<(N_NODES + 255) / 256, 256, 0, stream>>>(deg, dinv, N_NODES);

    // hl = h @ conv_lin_W[i]
    gemm_wmma_bf16<HID, false, false, false><<<gemmGridN, 256, 0, stream>>>(
        hin, nullptr, wl_bf + (size_t)i * HID * HID, nullptr, hl, HID);

    agg_init_kernel<<<(N_NODES * HID + 255) / 256, 256, 0, stream>>>(hl, dinv, agg,
                                                                     N_NODES * HID);
    edge_scatter_kernel<<<N_EDGES / 8, 256, 0, stream>>>(edge_index, wgate, dinv, hl, agg,
                                                         N_EDGES);

    // h = relu((agg + conv_bias[i]) @ gcn_W[i] + gcn_b[i])
    gemm_wmma_bf16<HID, true, true, true><<<gemmGridN, 256, 0, stream>>>(
        agg, conv_bias + (size_t)i * HID, wg_bf + (size_t)i * HID * HID,
        gcn_b + (size_t)i * HID, hbuf, HID);
    hin = hbuf;
  }

  // global mean pool -> pred_in[:, :128]
  fill_kernel<<<(G_GRAPHS * HID + 255) / 256, 256, 0, stream>>>(sums, 0.0f, G_GRAPHS * HID);
  fill_kernel<<<1, 256, 0, stream>>>(cnt, 0.0f, G_GRAPHS);
  pool_acc_kernel<<<N_NODES, HID, 0, stream>>>(hbuf, batch_idx, sums, cnt);
  pool_div_kernel<<<G_GRAPHS, HID, 0, stream>>>(sums, cnt, pred_in);

  // mol-feature MLP -> pred_in[:, 128:]
  linear_relu_kernel<<<G_GRAPHS, MH, 0, stream>>>(mol_feat, mlp_W0, mlp_b0, tmpA,
                                                  MIN_F, MH, MH, 0, 1);
  linear_relu_kernel<<<G_GRAPHS, MH, 0, stream>>>(tmpA, mlp_W1, mlp_b1, tmpB, MH, MH, MH, 0, 1);
  linear_relu_kernel<<<G_GRAPHS, MH, 0, stream>>>(tmpB, mlp_W2, mlp_b2, tmpA, MH, MH, MH, 0, 1);
  linear_relu_kernel<<<G_GRAPHS, MOUT, 0, stream>>>(tmpA, mlp_W3, mlp_b3, pred_in,
                                                    MH, MOUT, HID + MOUT, HID, 1);

  // predictor (WMMA): [256,256] -> [256,512] -> [256,512] -> [256,1]
  gemm_wmma_bf16<HID + MOUT, false, true, true><<<dim3(G_GRAPHS / 16, PH / 128), 256, 0, stream>>>(
      pred_in, nullptr, p0_bf, pred_b0, ph0, PH);
  gemm_wmma_bf16<PH, false, true, true><<<dim3(G_GRAPHS / 16, PH / 128), 256, 0, stream>>>(
      ph0, nullptr, p1_bf, pred_b1, ph1, PH);
  final_out_kernel<<<G_GRAPHS, 64, 0, stream>>>(ph1, out_W, out_b, out);
}
